// OriNetCRF_23364622090940
// MI455X (gfx1250) — matrix operand order repaired and checked by reading, not compiled, not run on hardware
//
#include <hip/hip_runtime.h>
#include <hip/hip_bf16.h>
#include <math.h>

#define DM 512
#define EM 1024
#define NS 16
#define NLY 8
#define LSEQ 4096
#define BATCH 2
#define MTOK (BATCH*LSEQ)      // 8192 tokens
#define NCH 64                 // scan chunks
#define CLEN 64                // chunk length (NCH*CLEN == LSEQ)

typedef __attribute__((ext_vector_type(16))) __bf16 v16bf;
typedef __attribute__((ext_vector_type(8)))  float  v8f;
typedef __attribute__((ext_vector_type(4)))  float  f32x4;
typedef __attribute__((ext_vector_type(4)))  int    v4i;

#define AS1 __attribute__((address_space(1)))
#define AS3 __attribute__((address_space(3)))

// ---------------- async global->LDS staging (CDNA5), with safe fallback ----------------
#if defined(__has_builtin)
#if __has_builtin(__builtin_amdgcn_global_load_async_to_lds_b128)
#define HAS_ASYNC_LDS 1
#endif
#endif

__device__ __forceinline__ void copy16_to_lds(__bf16* lds, const __bf16* g) {
#ifdef HAS_ASYNC_LDS
    __builtin_amdgcn_global_load_async_to_lds_b128((AS1 v4i*)g, (AS3 v4i*)lds, 0, 0);
#else
    *(f32x4*)lds = *(const f32x4*)g;
#endif
}

__device__ __forceinline__ void wait_async_le3() {
#ifdef HAS_ASYNC_LDS
#if defined(__has_builtin) && __has_builtin(__builtin_amdgcn_s_wait_asynccnt)
    __builtin_amdgcn_s_wait_asynccnt(3);
#else
    asm volatile("s_wait_asynccnt 3" ::: "memory");
#endif
#endif
}
__device__ __forceinline__ void wait_async_le0() {
#ifdef HAS_ASYNC_LDS
#if defined(__has_builtin) && __has_builtin(__builtin_amdgcn_s_wait_asynccnt)
    __builtin_amdgcn_s_wait_asynccnt(0);
#else
    asm volatile("s_wait_asynccnt 0" ::: "memory");
#endif
#endif
}

// ---------------- WMMA fragment loaders from LDS tiles (CDNA5 layouts, wave32) ----------
// tiles stored row-major [rows][32] bf16 (row stride 64B)
// A frag: row r0+(lane&15); j=0..7 -> K = 8*(lane>>4)+j ; j=8..15 -> +16
__device__ __forceinline__ v16bf lds_frag_a(const __bf16* base, int r0, int lane) {
    const __bf16* p = base + (size_t)(r0 + (lane & 15)) * 32 + ((lane >> 4) << 3);
    union { v16bf v; f32x4 q[2]; } u;
    u.q[0] = *(const f32x4*)(p);
    u.q[1] = *(const f32x4*)(p + 16);
    return u.v;
}
// B frag (from W[N,K] tile): col n = lane&15, j=0..7 -> K = 8*(lane>>4)+j ; j=8..15 -> +16
__device__ __forceinline__ v16bf lds_frag_b(const __bf16* base, int r0, int lane) {
    const __bf16* p = base + (size_t)(r0 + (lane & 15)) * 32 + ((lane >> 4) << 3);
    union { v16bf v; f32x4 q[2]; } u;
    u.q[0] = *(const f32x4*)(p);
    u.q[1] = *(const f32x4*)(p + 16);
    return u.v;
}
// C/D 16x16 f32: VGPR i -> row m0 + i + 8*(lane>>4), col n0 + (lane&15)
__device__ __forceinline__ void st_tile(float* C, int ldc, int m0, int n0,
                                        v8f acc, int lane, int accum) {
    int col = n0 + (lane & 15);
    int row0 = m0 + ((lane >> 4) << 3);
#pragma unroll
    for (int i = 0; i < 8; i++) {
        float* p = C + (size_t)(row0 + i) * ldc + col;
        if (accum) *p += acc[i]; else *p = acc[i];
    }
}

// ---------------- LDS-staged, double-buffered bf16 WMMA GEMM ----------------
// C[M,N] (+)= A[M,K] @ W[N,K]^T
// block = 256 threads (8 waves), block tile 128x64, wave tile 32x32 (2x2 WMMA tiles)
// per stage each thread issues 3 async 16B transfers (2 for A, 1 for B)
__global__ __launch_bounds__(256)
void gemm_wmma_bf16(const __bf16* __restrict__ A, const __bf16* __restrict__ W,
                    float* __restrict__ C, int M, int N, int K, int accum) {
    __shared__ __align__(16) __bf16 sA[2][128 * 32];
    __shared__ __align__(16) __bf16 sB[2][64 * 32];
    int tid = threadIdx.x;
    int lane = tid & 31;
    int wave = tid >> 5;
    int m0 = blockIdx.y * 128, n0 = blockIdx.x * 64;
    int wm = (wave >> 1) * 32;     // 0,32,64,96
    int wn = (wave & 1) * 32;      // 0,32

    int arow0 = (tid >> 2), ac = (tid & 3) * 8;        // A transfer 0: rows 0..63
    int arow1 = arow0 + 64;                            // A transfer 1: rows 64..127
    int brow = (tid >> 2), bc = (tid & 3) * 8;         // B: rows 0..63

    v8f a00 = {}, a01 = {}, a10 = {}, a11 = {};

    // prologue: stage 0
    {
        copy16_to_lds(sA[0] + arow0 * 32 + ac, A + (size_t)(m0 + arow0) * K + ac);
        copy16_to_lds(sA[0] + arow1 * 32 + ac, A + (size_t)(m0 + arow1) * K + ac);
        copy16_to_lds(sB[0] + brow * 32 + bc, W + (size_t)(n0 + brow) * K + bc);
    }
    int buf = 0;
    for (int k0 = 0; k0 < K; k0 += 32) {
        int kn = k0 + 32;
        bool more = kn < K;
        if (more) {
            int nb = buf ^ 1;
            copy16_to_lds(sA[nb] + arow0 * 32 + ac, A + (size_t)(m0 + arow0) * K + kn + ac);
            copy16_to_lds(sA[nb] + arow1 * 32 + ac, A + (size_t)(m0 + arow1) * K + kn + ac);
            copy16_to_lds(sB[nb] + brow * 32 + bc, W + (size_t)(n0 + brow) * K + kn + bc);
            wait_async_le3();          // previous stage drained, next still in flight
        } else {
            wait_async_le0();
        }
        __syncthreads();               // LDS tiles visible to all waves
        v16bf fa0 = lds_frag_a(sA[buf], wm, lane);
        v16bf fa1 = lds_frag_a(sA[buf], wm + 16, lane);
        v16bf fb0 = lds_frag_b(sB[buf], wn, lane);
        v16bf fb1 = lds_frag_b(sB[buf], wn + 16, lane);
        a00 = __builtin_amdgcn_wmma_f32_16x16x32_bf16(false, fa0, false, fb0, (short)0, a00, false, false);
        a01 = __builtin_amdgcn_wmma_f32_16x16x32_bf16(false, fa0, false, fb1, (short)0, a01, false, false);
        a10 = __builtin_amdgcn_wmma_f32_16x16x32_bf16(false, fa1, false, fb0, (short)0, a10, false, false);
        a11 = __builtin_amdgcn_wmma_f32_16x16x32_bf16(false, fa1, false, fb1, (short)0, a11, false, false);
        __syncthreads();               // all waves done reading buf before re-fill
        buf ^= 1;
    }
    st_tile(C, N, m0 + wm,      n0 + wn,      a00, lane, accum);
    st_tile(C, N, m0 + wm,      n0 + wn + 16, a01, lane, accum);
    st_tile(C, N, m0 + wm + 16, n0 + wn,      a10, lane, accum);
    st_tile(C, N, m0 + wm + 16, n0 + wn + 16, a11, lane, accum);
}

// ---------------- Elementwise / small kernels ----------------
__global__ void embed_gather(const int* __restrict__ ids, const float* __restrict__ emb,
                             float* __restrict__ h) {
    int row = blockIdx.x;
    int id = ids[row];
    const float* s = emb + (size_t)id * DM;
    float* d = h + (size_t)row * DM;
    for (int c = threadIdx.x; c < DM; c += blockDim.x) d[c] = s[c];
}

__global__ void cvt_bf16(const float* __restrict__ s, __bf16* __restrict__ d, long n) {
    long i = (long)blockIdx.x * blockDim.x + threadIdx.x;
    if (i < n) d[i] = (__bf16)s[i];
}

// take cols 0..31 of [MTOK,64] fp32 -> [MTOK,32] bf16 (dt input for W_dt GEMM)
__global__ void slice_bf16(const float* __restrict__ s, __bf16* __restrict__ d) {
    int row = blockIdx.x; int c = threadIdx.x;  // 32 threads
    d[(size_t)row * 32 + c] = (__bf16)s[(size_t)row * 64 + c];
}

// rmsnorm over D=512; one block (128 threads) per row; bf16 and/or f32 out
__global__ __launch_bounds__(128)
void rmsnorm_k(const float* __restrict__ X, const float* __restrict__ w,
               __bf16* __restrict__ Yb, float* __restrict__ Yf) {
    int row = blockIdx.x;
    const float* x = X + (size_t)row * DM;
    float v[4]; float ss = 0.f;
#pragma unroll
    for (int i = 0; i < 4; i++) { v[i] = x[threadIdx.x + i * 128]; ss += v[i] * v[i]; }
    __shared__ float red[4];
    for (int o = 16; o > 0; o >>= 1) ss += __shfl_down(ss, o, 32);
    if ((threadIdx.x & 31) == 0) red[threadIdx.x >> 5] = ss;
    __syncthreads();
    ss = red[0] + red[1] + red[2] + red[3];
    float r = rsqrtf(ss * (1.0f / DM) + 1e-5f);
#pragma unroll
    for (int i = 0; i < 4; i++) {
        int c = threadIdx.x + i * 128;
        float y = v[i] * r * w[c];
        if (Yb) Yb[(size_t)row * DM + c] = (__bf16)y;
        if (Yf) Yf[(size_t)row * DM + c] = y;
    }
}

// causal (fwd) / anti-causal-with-reversed-taps (rev) depthwise conv K=4 + bias + silu
__global__ void conv_silu(const float* __restrict__ xz, const float* __restrict__ cw,
                          const float* __restrict__ cb, float* __restrict__ uf,
                          __bf16* __restrict__ ub, int rev) {
    long idx = (long)blockIdx.x * blockDim.x + threadIdx.x;   // over MTOK*EM
    int e = (int)(idx & (EM - 1));
    int row = (int)(idx >> 10);
    int b = row >> 12, l = row & (LSEQ - 1);
    float acc = cb[e];
    if (!rev) {
#pragma unroll
        for (int k = 0; k < 4; k++) {
            int li = l - 3 + k;
            if (li >= 0) acc += cw[e * 4 + k] * xz[((size_t)(b * LSEQ + li)) * 2048 + e];
        }
    } else {
#pragma unroll
        for (int k = 0; k < 4; k++) {
            int li = l + k;
            if (li < LSEQ) acc += cw[e * 4 + (3 - k)] * xz[((size_t)(b * LSEQ + li)) * 2048 + e];
        }
    }
    float s = acc / (1.f + __expf(-acc));
    uf[idx] = s;
    ub[idx] = (__bf16)s;
}

__global__ void dt_softplus(float* __restrict__ dt, const float* __restrict__ bdt) {
    long idx = (long)blockIdx.x * blockDim.x + threadIdx.x;   // over MTOK*EM
    int e = (int)(idx & (EM - 1));
    float x = dt[idx] + bdt[e];
    dt[idx] = (x > 20.f) ? x : log1pf(__expf(x));
}

// -------- chunked selective scan --------
// pass1: per-chunk transition product P and local state S (h0 = 0)
__global__ __launch_bounds__(256)
void scan_pass1(const float* __restrict__ ucf, const float* __restrict__ dtf,
                const float* __restrict__ dbc, const float* __restrict__ Alog,
                float* __restrict__ P, float* __restrict__ S, int rev) {
    __shared__ float sB[CLEN][NS];
    int tid = threadIdx.x;
    int c = blockIdx.x;                      // chunk
    int ch = blockIdx.y * 256 + tid;         // 0..2047 : b = ch>>10, e = ch&1023
    int b = ch >> 10, e = ch & (EM - 1);
    float An[NS];
#pragma unroll
    for (int n = 0; n < NS; n++) An[n] = -__expf(Alog[(size_t)e * NS + n]);
    for (int i = tid; i < CLEN * NS; i += 256) {
        int j = i >> 4, n = i & 15;
        int g = c * CLEN + j;
        int l = rev ? (LSEQ - 1 - g) : g;
        sB[j][n] = dbc[((size_t)(b * LSEQ + l)) * 64 + 32 + n];
    }
    __syncthreads();
    float p[NS], h[NS];
#pragma unroll
    for (int n = 0; n < NS; n++) { p[n] = 1.f; h[n] = 0.f; }
    for (int j = 0; j < CLEN; j++) {
        int g = c * CLEN + j;
        int l = rev ? (LSEQ - 1 - g) : g;
        size_t re = ((size_t)(b * LSEQ + l)) * EM + e;
        float dtv = dtf[re], uv = ucf[re];
        float du = dtv * uv;
#pragma unroll
        for (int n = 0; n < NS; n++) {
            float dA = __expf(dtv * An[n]);
            p[n] *= dA;
            h[n] = dA * h[n] + du * sB[j][n];
        }
    }
    size_t o = ((size_t)c * 2048 + ch) * NS;
#pragma unroll
    for (int n = 0; n < NS; n++) { P[o + n] = p[n]; S[o + n] = h[n]; }
}

// sequential combine over chunks: H[c] = state entering chunk c
__global__ void scan_combine(const float* __restrict__ P, const float* __restrict__ S,
                             float* __restrict__ H) {
    int idx = blockIdx.x * 256 + threadIdx.x;    // 0..2048*16-1
    float hh = 0.f;
    for (int c = 0; c < NCH; c++) {
        size_t o = (size_t)c * (2048 * NS) + idx;
        H[o] = hh;
        hh = P[o] * hh + S[o];
    }
}

// pass2: re-scan chunk with correct initial state, emit y (+ D skip)
__global__ __launch_bounds__(256)
void scan_pass2(const float* __restrict__ ucf, const float* __restrict__ dtf,
                const float* __restrict__ dbc, const float* __restrict__ Alog,
                const float* __restrict__ Dp, const float* __restrict__ H,
                float* __restrict__ y, int rev) {
    __shared__ float sB[CLEN][NS];
    __shared__ float sC[CLEN][NS];
    int tid = threadIdx.x;
    int c = blockIdx.x;
    int ch = blockIdx.y * 256 + tid;
    int b = ch >> 10, e = ch & (EM - 1);
    float An[NS];
#pragma unroll
    for (int n = 0; n < NS; n++) An[n] = -__expf(Alog[(size_t)e * NS + n]);
    for (int i = tid; i < CLEN * 32; i += 256) {
        int j = i >> 5, t = i & 31;
        int g = c * CLEN + j;
        int l = rev ? (LSEQ - 1 - g) : g;
        float v = dbc[((size_t)(b * LSEQ + l)) * 64 + 32 + t];
        if (t < 16) sB[j][t] = v; else sC[j][t - 16] = v;
    }
    __syncthreads();
    float h[NS];
    size_t ho = ((size_t)c * 2048 + ch) * NS;
#pragma unroll
    for (int n = 0; n < NS; n++) h[n] = H[ho + n];
    float dsk = Dp[e];
    for (int j = 0; j < CLEN; j++) {
        int g = c * CLEN + j;
        int l = rev ? (LSEQ - 1 - g) : g;
        size_t re = ((size_t)(b * LSEQ + l)) * EM + e;
        float dtv = dtf[re], uv = ucf[re];
        float du = dtv * uv;
        float yv = 0.f;
#pragma unroll
        for (int n = 0; n < NS; n++) {
            float dA = __expf(dtv * An[n]);
            h[n] = dA * h[n] + du * sB[j][n];
            yv += h[n] * sC[j][n];
        }
        y[re] = yv + uv * dsk;
    }
}

__global__ void gate_bf16(const float* __restrict__ y, const float* __restrict__ xz,
                          __bf16* __restrict__ g) {
    long idx = (long)blockIdx.x * blockDim.x + threadIdx.x;   // over MTOK*EM
    int e = (int)(idx & (EM - 1));
    int row = (int)(idx >> 10);
    float z = xz[(size_t)row * 2048 + EM + e];
    float s = z / (1.f + __expf(-z));
    g[idx] = (__bf16)(y[idx] * s);
}

__global__ void head_logits(const float* __restrict__ hn, const float* __restrict__ hw,
                            float* __restrict__ out) {
    int row = blockIdx.x * blockDim.x + threadIdx.x;
    if (row >= MTOK) return;
    const float* x = hn + (size_t)row * DM;
    float a0 = 0.f, a1 = 0.f;
    for (int d = 0; d < DM; d++) { float v = x[d]; a0 += v * hw[d]; a1 += v * hw[DM + d]; }
    out[row * 2 + 0] = a0; out[row * 2 + 1] = a1;
}

__device__ __forceinline__ float lse2(float a, float b) {
    float m = fmaxf(a, b);
    return m + logf(__expf(a - m) + __expf(b - m));
}

__global__ void crf_nll(const float* __restrict__ lg_all, const int* __restrict__ lab_all,
                        const float* __restrict__ st, const float* __restrict__ tr,
                        const float* __restrict__ en, float* __restrict__ loss) {
    if (threadIdx.x != 0) return;
    int b = blockIdx.x;
    const float* lg = lg_all + (size_t)b * LSEQ * 2;
    const int* lab = lab_all + (size_t)b * LSEQ;
    float num = st[lab[0]];
    for (int l = 0; l < LSEQ; l++) num += lg[l * 2 + lab[l]];
    for (int l = 0; l + 1 < LSEQ; l++) num += tr[lab[l] * 2 + lab[l + 1]];
    num += en[lab[LSEQ - 1]];
    float a0 = st[0] + lg[0], a1 = st[1] + lg[1];
    for (int l = 1; l < LSEQ; l++) {
        float e0 = lg[l * 2], e1 = lg[l * 2 + 1];
        float n0 = lse2(a0 + tr[0], a1 + tr[2]) + e0;
        float n1 = lse2(a0 + tr[1], a1 + tr[3]) + e1;
        a0 = n0; a1 = n1;
    }
    loss[b] = lse2(a0 + en[0], a1 + en[1]) - num;
}

// ---------------- host ----------------
extern "C" void kernel_launch(void* const* d_in, const int* in_sizes, int n_in,
                              void* d_out, int out_size, void* d_ws, size_t ws_size,
                              hipStream_t stream) {
    const int*   input_ids = (const int*)  d_in[0];
    const int*   labels    = (const int*)  d_in[1];
    const float* embed     = (const float*)d_in[2];
    const float* W_in      = (const float*)d_in[3];
    const float* conv_w    = (const float*)d_in[4];
    const float* conv_b    = (const float*)d_in[5];
    const float* W_x       = (const float*)d_in[6];
    const float* W_dt      = (const float*)d_in[7];
    const float* b_dt      = (const float*)d_in[8];
    const float* A_log     = (const float*)d_in[9];
    const float* D_skip    = (const float*)d_in[10];
    const float* W_out     = (const float*)d_in[11];
    const float* norm_w    = (const float*)d_in[12];
    const float* norm_f    = (const float*)d_in[13];
    const float* head_w    = (const float*)d_in[14];
    const float* crf_start = (const float*)d_in[15];
    const float* crf_trans = (const float*)d_in[16];
    const float* crf_end   = (const float*)d_in[17];
    float* out = (float*)d_out;

    char* base = (char*)d_ws;
    size_t off = 0;
    auto alloc = [&](size_t bytes) -> void* {
        off = (off + 255) & ~(size_t)255;
        void* p = base + off; off += bytes; return p;
    };
    float*  hbuf  = (float*) alloc((size_t)MTOK * DM * 4);
    __bf16* xnb   = (__bf16*)alloc((size_t)MTOK * DM * 2);
    float*  xz    = (float*) alloc((size_t)MTOK * 2048 * 4);
    float*  ucf   = (float*) alloc((size_t)MTOK * EM * 4);
    __bf16* ucb   = (__bf16*)alloc((size_t)MTOK * EM * 2);
    float*  dbc   = (float*) alloc((size_t)MTOK * 64 * 4);
    __bf16* dtinb = (__bf16*)alloc((size_t)MTOK * 32 * 2);
    float*  dtf   = (float*) alloc((size_t)MTOK * EM * 4);
    float*  ybuf  = (float*) alloc((size_t)MTOK * EM * 4);
    __bf16* gbuf  = (__bf16*)alloc((size_t)MTOK * EM * 2);
    float*  hn    = (float*) alloc((size_t)MTOK * DM * 4);
    float*  Pb    = (float*) alloc((size_t)NCH * 2048 * NS * 4);
    float*  Sb    = (float*) alloc((size_t)NCH * 2048 * NS * 4);
    float*  Hb    = (float*) alloc((size_t)NCH * 2048 * NS * 4);
    __bf16* WinB  = (__bf16*)alloc((size_t)NLY * 2048 * DM * 2);
    __bf16* WxB   = (__bf16*)alloc((size_t)NLY * 64 * EM * 2);
    __bf16* WdtB  = (__bf16*)alloc((size_t)NLY * EM * 32 * 2);
    __bf16* WoutB = (__bf16*)alloc((size_t)NLY * DM * EM * 2);

    auto g1 = [](long n) { return (unsigned)((n + 255) / 256); };
    cvt_bf16<<<g1((long)NLY*2048*DM), 256, 0, stream>>>(W_in,  WinB,  (long)NLY*2048*DM);
    cvt_bf16<<<g1((long)NLY*64*EM),   256, 0, stream>>>(W_x,   WxB,   (long)NLY*64*EM);
    cvt_bf16<<<g1((long)NLY*EM*32),   256, 0, stream>>>(W_dt,  WdtB,  (long)NLY*EM*32);
    cvt_bf16<<<g1((long)NLY*DM*EM),   256, 0, stream>>>(W_out, WoutB, (long)NLY*DM*EM);

    embed_gather<<<MTOK, 128, 0, stream>>>(input_ids, embed, hbuf);

    const long nEW = (long)MTOK * EM;          // 8.4M elementwise threads
    for (int l = 0; l < NLY; l++) {
        rmsnorm_k<<<MTOK, 128, 0, stream>>>(hbuf, norm_w + (size_t)l * DM, xnb, nullptr);
        // xz = xn @ W_in^T  [8192 x 2048]
        gemm_wmma_bf16<<<dim3(2048/64, MTOK/128), 256, 0, stream>>>(
            xnb, WinB + (size_t)l * 2048 * DM, xz, MTOK, 2048, DM, 0);
        for (int dir = 0; dir < 2; dir++) {
            conv_silu<<<g1(nEW), 256, 0, stream>>>(xz, conv_w + (size_t)l * EM * 4,
                                                   conv_b + (size_t)l * EM, ucf, ucb, dir);
            // dbc = uc @ W_x^T  [8192 x 64]
            gemm_wmma_bf16<<<dim3(64/64, MTOK/128), 256, 0, stream>>>(
                ucb, WxB + (size_t)l * 64 * EM, dbc, MTOK, 64, EM, 0);
            slice_bf16<<<MTOK, 32, 0, stream>>>(dbc, dtinb);
            // dt_raw = dt_in @ W_dt^T  [8192 x 1024]
            gemm_wmma_bf16<<<dim3(EM/64, MTOK/128), 256, 0, stream>>>(
                dtinb, WdtB + (size_t)l * EM * 32, dtf, MTOK, EM, 32, 0);
            dt_softplus<<<g1(nEW), 256, 0, stream>>>(dtf, b_dt + (size_t)l * EM);
            scan_pass1<<<dim3(NCH, 8), 256, 0, stream>>>(
                ucf, dtf, dbc, A_log + (size_t)l * EM * NS, Pb, Sb, dir);
            scan_combine<<<(2048 * NS) / 256, 256, 0, stream>>>(Pb, Sb, Hb);
            scan_pass2<<<dim3(NCH, 8), 256, 0, stream>>>(
                ucf, dtf, dbc, A_log + (size_t)l * EM * NS,
                D_skip + (size_t)l * EM, Hb, ybuf, dir);
            gate_bf16<<<g1(nEW), 256, 0, stream>>>(ybuf, xz, gbuf);
            // h += (y * silu(z)) @ W_out^T  [8192 x 512], beta = 1
            gemm_wmma_bf16<<<dim3(DM/64, MTOK/128), 256, 0, stream>>>(
                gbuf, WoutB + (size_t)l * DM * EM, hbuf, MTOK, DM, EM, 1);
        }
    }
    rmsnorm_k<<<MTOK, 128, 0, stream>>>(hbuf, norm_f, nullptr, hn);
    head_logits<<<(MTOK + 127) / 128, 128, 0, stream>>>(hn, head_w, out);
    crf_nll<<<BATCH, 32, 0, stream>>>(out, labels, crf_start, crf_trans, crf_end,
                                      out + (size_t)MTOK * 2);
}
